// MixtureOfExperts_65807488910001
// MI455X (gfx1250) — matrix operand order repaired and checked by reading, not compiled, not run on hardware
//
#include <hip/hip_runtime.h>
#include <hip/hip_bf16.h>

// ---------------------------------------------------------------------------
// MoE (top-2 of 8 experts) + LayerNorm for MI455X (gfx1250, wave32, WMMA).
// B=8 T=2048 D=1024 E=8 F=4096.  bf16 WMMA, f32 accumulate.
// v3: fix async-to-LDS builtin argument types (int __vector(4) AS1/AS3 ptrs).
// ---------------------------------------------------------------------------

#define NTOK   16384      // B*T
#define DDIM   1024
#define FDIM   4096
#define NEXP   8
#define CAP    16384      // worst-case tokens per expert

typedef __attribute__((ext_vector_type(16))) __bf16 v16bf;
typedef __attribute__((ext_vector_type(8)))  float  v8f;

union V16 { v16bf v; unsigned int u[8]; uint4 q[2]; };

static __device__ __forceinline__ unsigned short f2bf(float f) {
    unsigned u = __builtin_bit_cast(unsigned, f);
    unsigned r = (u + 0x7FFFu + ((u >> 16) & 1u)) >> 16;   // round-nearest-even
    return (unsigned short)r;
}
static __device__ __forceinline__ unsigned pack2(float a, float b) {
    return (unsigned)f2bf(a) | ((unsigned)f2bf(b) << 16);
}

#if defined(__gfx1250__) && __has_builtin(__builtin_amdgcn_global_load_async_to_lds_b128)
#define HAVE_ASYNC_LDS 1
// Builtin param types (from hipcc diagnostic): int __vector(4) in AS1 / AS3.
typedef int v4i_ls __attribute__((vector_size(16)));
typedef __attribute__((address_space(1))) v4i_ls GAS_v4i;
typedef __attribute__((address_space(3))) v4i_ls LAS_v4i;
#endif

// ---------------------------------------------------------------------------
// Kernel 0: f32 -> bf16 bulk conversion (n multiple of 2048; 8 elems/thread).
// ---------------------------------------------------------------------------
__global__ __launch_bounds__(256)
void cvt_bf16_kernel(const float* __restrict__ src, unsigned short* __restrict__ dst) {
    size_t i = ((size_t)blockIdx.x * 256 + threadIdx.x) * 8;
    float4 a = *(const float4*)(src + i);
    float4 b = *(const float4*)(src + i + 4);
    uint4 o;
    o.x = pack2(a.x, a.y); o.y = pack2(a.z, a.w);
    o.z = pack2(b.x, b.y); o.w = pack2(b.z, b.w);
    *(uint4*)(dst + i) = o;
}

// ---------------------------------------------------------------------------
// Kernel 1: router logits + top-2 softmax + per-expert compaction.
// One wave (32 lanes) per token.  grid = NTOK/8 blocks of 256.
// ---------------------------------------------------------------------------
__global__ __launch_bounds__(256)
void router_kernel(const float* __restrict__ x, const float* __restrict__ rw,
                   float* __restrict__ logits_out,
                   int* __restrict__ cnt, int* __restrict__ tokidx,
                   float* __restrict__ gates) {
    const int wid  = threadIdx.x >> 5;
    const int lane = threadIdx.x & 31;
    const int token = blockIdx.x * 8 + wid;

    const float* xr = x + (size_t)token * DDIM;
    float xv[32];
#pragma unroll
    for (int i = 0; i < 32; ++i) xv[i] = xr[lane + 32 * i];

    float lg[NEXP];
#pragma unroll
    for (int e = 0; e < NEXP; ++e) {
        const float* w = rw + e * DDIM;
        float acc = 0.f;
#pragma unroll
        for (int i = 0; i < 32; ++i) acc += xv[i] * w[lane + 32 * i];
#pragma unroll
        for (int s = 16; s > 0; s >>= 1) acc += __shfl_xor(acc, s);
        lg[e] = acc;                       // butterfly: all lanes hold result
    }

    if (lane == 0) {
        float* lo = logits_out + (size_t)token * NEXP;
#pragma unroll
        for (int e = 0; e < NEXP; ++e) lo[e] = lg[e];

        int   i1 = 0; float v1 = lg[0];
#pragma unroll
        for (int e = 1; e < NEXP; ++e) if (lg[e] > v1) { v1 = lg[e]; i1 = e; }
        int   i2 = -1; float v2 = -__builtin_inff();
#pragma unroll
        for (int e = 0; e < NEXP; ++e)
            if (e != i1 && lg[e] > v2) { v2 = lg[e]; i2 = e; }

        float e2 = __expf(v2 - v1);
        float p1 = 1.f / (1.f + e2);
        float p2 = e2 / (1.f + e2);

        int s1 = atomicAdd(&cnt[i1], 1);
        tokidx[i1 * CAP + s1] = token;  gates[i1 * CAP + s1] = p1;
        int s2 = atomicAdd(&cnt[i2], 1);
        tokidx[i2 * CAP + s2] = token;  gates[i2 * CAP + s2] = p2;
    }
}

// ---------------------------------------------------------------------------
// Kernel 2: per-expert FFN, bf16 WMMA.  grid = (CAP/16, NEXP), block = 256.
// Block handles a tile of 16 routed tokens for expert blockIdx.y.
// All operands pre-converted bf16: fragments are 2x16B loads each.
// ---------------------------------------------------------------------------
__global__ __launch_bounds__(256)
void expert_ffn_kernel(const unsigned short* __restrict__ xbf,
                       const unsigned short* __restrict__ wfcbf,
                       const float* __restrict__ bfc,
                       const unsigned short* __restrict__ wprojbf,
                       const float* __restrict__ bproj,
                       const int* __restrict__ cnt, const int* __restrict__ tokidx,
                       const float* __restrict__ gates,
                       float* __restrict__ ybuf) {
    const int e    = blockIdx.y;
    const int tile = blockIdx.x;
    const int n    = cnt[e];
    if (tile * 16 >= n) return;                    // uniform across block

    __shared__ unsigned short sx[16 * DDIM];       // x tile, bf16 (32 KB)
    __shared__ unsigned short sh[16 * 128];        // h chunk, bf16 (4 KB)
    __shared__ int   stok[16];
    __shared__ float sgate[16];

    const int tid = threadIdx.x;
    if (tid < 16) {
        int a = tile * 16 + tid;
        if (a < n) { stok[tid] = tokidx[e * CAP + a]; sgate[tid] = gates[e * CAP + a]; }
        else       { stok[tid] = -1;                  sgate[tid] = 0.f; }
    }
    __syncthreads();

    // Stage 16 bf16 token rows into LDS: 16 threads/row, 128 B (8x b128) each.
    {
        const int r = tid >> 4, seg = tid & 15;
        const int tok = stok[r];
        unsigned short* lp = &sx[r * DDIM + seg * 64];
        if (tok >= 0) {
            const unsigned short* gp = xbf + (size_t)tok * DDIM + seg * 64;
#ifdef HAVE_ASYNC_LDS
            GAS_v4i* gq = (GAS_v4i*)(uintptr_t)gp;
            LAS_v4i* lq = (LAS_v4i*)(uintptr_t)lp;
#pragma unroll
            for (int i = 0; i < 8; ++i)
                __builtin_amdgcn_global_load_async_to_lds_b128(gq + i, lq + i, 0, 0);
#else
            const uint4* gq = (const uint4*)gp;
            uint4* lq = (uint4*)lp;
#pragma unroll
            for (int i = 0; i < 8; ++i) lq[i] = gq[i];
#endif
        } else {
            uint4* lq = (uint4*)lp;
            const uint4 z = {0u, 0u, 0u, 0u};
#pragma unroll
            for (int i = 0; i < 8; ++i) lq[i] = z;
        }
    }
#ifdef HAVE_ASYNC_LDS
#if __has_builtin(__builtin_amdgcn_s_wait_asynccnt)
    __builtin_amdgcn_s_wait_asynccnt(0);
#else
    asm volatile("s_wait_asynccnt 0" ::: "memory");
#endif
#endif
    __syncthreads();

    const int wave  = tid >> 5;
    const int lane  = tid & 31;
    const int l15   = lane & 15;
    const int lhalf = (lane & 16) ? 1 : 0;
    const int aKb   = lhalf ? 16 : 0;     // A-fragment K base (contiguous k = 16h + s)
    const int bKb   = lhalf ? 16 : 0;     // B-fragment K base

    v8f acc[8];
#pragma unroll
    for (int t = 0; t < 8; ++t) acc[t] = (v8f){0.f,0.f,0.f,0.f,0.f,0.f,0.f,0.f};

    for (int fc = 0; fc < FDIM / 128; ++fc) {
        const int f0 = fc * 128 + wave * 16;       // this wave's 16 f-columns

        // ---- phase 1: h[16 x 16] = x_tile @ wfc[e, f0:f0+16, :]^T ----------
        v8f h = (v8f){0.f,0.f,0.f,0.f,0.f,0.f,0.f,0.f};
        const unsigned short* Bb =
            wfcbf + ((size_t)e * FDIM + f0 + l15) * DDIM + bKb;
        const unsigned short* Ab = &sx[l15 * DDIM + aKb];
        for (int ks = 0; ks < DDIM / 32; ++ks) {
            V16 A, Bf;
            A.q[0]  = *(const uint4*)(Ab + ks * 32);
            A.q[1]  = *(const uint4*)(Ab + ks * 32 + 8);
            Bf.q[0] = *(const uint4*)(Bb + ks * 32);
            Bf.q[1] = *(const uint4*)(Bb + ks * 32 + 8);
            h = __builtin_amdgcn_wmma_f32_16x16x32_bf16(
                    false, A.v, false, Bf.v, (short)0, h, false, false);
        }

        // bias + exact GELU, write bf16 h chunk to LDS
        const float bias = bfc[e * FDIM + f0 + l15];
#pragma unroll
        for (int j = 0; j < 8; ++j) {
            float v = h[j] + bias;
            v = 0.5f * v * (1.0f + erff(v * 0.70710678118654752f));
            int M = j + lhalf * 8;
            sh[M * 128 + wave * 16 + l15] = f2bf(v);
        }
        __syncthreads();

        // ---- phase 2: y += h_chunk[16 x 128] @ wproj[e, :, fchunk]^T -------
#pragma unroll
        for (int ks = 0; ks < 4; ++ks) {
            V16 A;
            A.q[0] = *(const uint4*)&sh[l15 * 128 + ks * 32 + aKb];
            A.q[1] = *(const uint4*)&sh[l15 * 128 + ks * 32 + aKb + 8];
#pragma unroll
            for (int t = 0; t < 8; ++t) {
                const int drow = (wave + 8 * t) * 16 + l15;
                const unsigned short* bp =
                    wprojbf + ((size_t)e * DDIM + drow) * FDIM
                    + fc * 128 + ks * 32 + bKb;
                V16 Bf;
                Bf.q[0] = *(const uint4*)bp;
                Bf.q[1] = *(const uint4*)(bp + 8);
                acc[t] = __builtin_amdgcn_wmma_f32_16x16x32_bf16(
                            false, A.v, false, Bf.v, (short)0, acc[t], false, false);
            }
        }
        __syncthreads();
    }

    // ---- epilogue: bias, gate, scatter-add into pre-LN buffer --------------
#pragma unroll
    for (int t = 0; t < 8; ++t) {
        const int d = (wave + 8 * t) * 16 + l15;
        const float bp = bproj[e * DDIM + d];
#pragma unroll
        for (int j = 0; j < 8; ++j) {
            const int M = j + lhalf * 8;
            const int tok = stok[M];
            if (tok >= 0) {
                float val = (acc[t][j] + bp) * sgate[M];
                atomicAdd(&ybuf[(size_t)tok * DDIM + d], val);
            }
        }
    }
}

// ---------------------------------------------------------------------------
// Kernel 3: LayerNorm over D.  One block (256 threads) per token.
// ---------------------------------------------------------------------------
__global__ __launch_bounds__(256)
void ln_kernel(const float* __restrict__ ybuf, const float* __restrict__ g,
               const float* __restrict__ b, float* __restrict__ out) {
    const int token = blockIdx.x;
    const int tid = threadIdx.x;
    const float* row = ybuf + (size_t)token * DDIM;

    float v[4], s = 0.f, s2 = 0.f;
#pragma unroll
    for (int i = 0; i < 4; ++i) {
        float t = row[tid + 256 * i];
        v[i] = t; s += t; s2 += t * t;
    }
#pragma unroll
    for (int m = 16; m > 0; m >>= 1) { s += __shfl_xor(s, m); s2 += __shfl_xor(s2, m); }

    __shared__ float ws1[8], ws2[8];
    const int wave = tid >> 5, lane = tid & 31;
    if (lane == 0) { ws1[wave] = s; ws2[wave] = s2; }
    __syncthreads();
    s = 0.f; s2 = 0.f;
#pragma unroll
    for (int i = 0; i < 8; ++i) { s += ws1[i]; s2 += ws2[i]; }

    const float mu  = s * (1.f / DDIM);
    const float var = s2 * (1.f / DDIM) - mu * mu;
    const float inv = rsqrtf(var + 1e-5f);
#pragma unroll
    for (int i = 0; i < 4; ++i) {
        int d = tid + 256 * i;
        out[(size_t)token * DDIM + d] = (v[i] - mu) * inv * g[d] + b[d];
    }
}

// ---------------------------------------------------------------------------
extern "C" void kernel_launch(void* const* d_in, const int* in_sizes, int n_in,
                              void* d_out, int out_size, void* d_ws, size_t ws_size,
                              hipStream_t stream) {
    const float* x        = (const float*)d_in[0];
    const float* router_w = (const float*)d_in[1];
    const float* wfc      = (const float*)d_in[2];
    const float* bfc      = (const float*)d_in[3];
    const float* wproj    = (const float*)d_in[4];
    const float* bproj    = (const float*)d_in[5];
    const float* ln_g     = (const float*)d_in[6];
    const float* ln_b     = (const float*)d_in[7];
    // d_in[8] = top_k (fixed at 2)

    float* out    = (float*)d_out;
    float* logits = out + (size_t)NTOK * DDIM;

    // Workspace layout (16B-aligned sections):
    //   ybuf   : NTOK*DDIM f32          (64 MB)
    //   xbf    : NTOK*DDIM bf16         (32 MB)
    //   wfcbf  : E*F*D bf16             (64 MB)
    //   wprojbf: E*D*F bf16             (64 MB)
    //   cnt / tokidx / gates            (~1 MB)
    char*  ws      = (char*)d_ws;
    float* ybuf    = (float*)ws;
    unsigned short* xbf     = (unsigned short*)(ws + (size_t)NTOK * DDIM * 4);
    unsigned short* wfcbf   = xbf + (size_t)NTOK * DDIM;
    unsigned short* wprojbf = wfcbf + (size_t)NEXP * FDIM * DDIM;
    int*   cnt     = (int*)(wprojbf + (size_t)NEXP * DDIM * FDIM);
    int*   tokidx  = cnt + 16;
    float* gates   = (float*)(tokidx + NEXP * CAP);

    (void)hipMemsetAsync(ybuf, 0, (size_t)NTOK * DDIM * sizeof(float), stream);
    (void)hipMemsetAsync(cnt, 0, 16 * sizeof(int), stream);

    // bf16 pre-conversion (8 elements per thread, exact grids)
    cvt_bf16_kernel<<<(NTOK * (size_t)DDIM) / 2048, 256, 0, stream>>>(x, xbf);
    cvt_bf16_kernel<<<((size_t)NEXP * FDIM * DDIM) / 2048, 256, 0, stream>>>(wfc, wfcbf);
    cvt_bf16_kernel<<<((size_t)NEXP * DDIM * FDIM) / 2048, 256, 0, stream>>>(wproj, wprojbf);

    router_kernel<<<NTOK / 8, 256, 0, stream>>>(x, router_w, logits,
                                                cnt, tokidx, gates);

    dim3 grid(CAP / 16, NEXP);
    expert_ffn_kernel<<<grid, 256, 0, stream>>>(xbf, wfcbf, bfc, wprojbf, bproj,
                                                cnt, tokidx, gates, ybuf);

    ln_kernel<<<NTOK, 256, 0, stream>>>(ybuf, ln_g, ln_b, out);
}
